// TxGNN_76802605187462
// MI455X (gfx1250) — matrix operand order repaired and checked by reading, not compiled, not run on hardware
//
#include <hip/hip_runtime.h>
#include <hip/hip_bf16.h>

// ---------------------------------------------------------------------------
// EdgeSAGE 2-layer GNN for MI455X (gfx1250), bf16 WMMA with f32 accumulate.
// ---------------------------------------------------------------------------

typedef __bf16 bf16_t;
typedef __attribute__((ext_vector_type(16))) __bf16 v16bf;
typedef __attribute__((ext_vector_type(8)))  __bf16 v8bf;
typedef __attribute__((ext_vector_type(8)))  float  v8f;

#define DFEAT   128   // D_IN == D_HID == D_OUT
#define DEDGE   64
#define KMSG    (DFEAT + DEDGE)   // 192
#define KUPD    (DFEAT + DFEAT)   // 256
#define TILE_M  64                // edges / nodes per workgroup
#define TPB     256               // 8 wave32 -> 8 N-tiles of 16 = 128 outputs

union AFrag { v16bf v; v8bf h[2]; };

// ---------------------------------------------------------------------------
// Weight transpose + f32->bf16: WT[n*K + k] = (bf16) W[k*N + n],  N == 128
// ---------------------------------------------------------------------------
__global__ void transpose_w_kernel(const float* __restrict__ W,
                                   bf16_t* __restrict__ WT, int K) {
    int idx = blockIdx.x * blockDim.x + threadIdx.x;
    if (idx < K * DFEAT) {
        int n = idx / K;
        int k = idx - n * K;
        WT[idx] = (bf16_t)W[k * DFEAT + n];
    }
}

__global__ void zero_kernel(float* __restrict__ p, long long n) {
    long long i = (long long)blockIdx.x * blockDim.x + threadIdx.x;
    if (i < n) p[i] = 0.0f;
}

// ---------------------------------------------------------------------------
// Message + scatter:  m = relu([x[src] , eattr] @ Wm + bm); aggr[dst] += m
// A tile: 64 edges x 192 (bf16 in LDS), B: WT [128 x 192] bf16 (L2 resident).
// ---------------------------------------------------------------------------
__global__ __launch_bounds__(TPB)
void msg_scatter_kernel(const float* __restrict__ x,      // [N, 128]
                        const float* __restrict__ eattr,  // [E, 64]
                        const int*   __restrict__ srcIdx,
                        const int*   __restrict__ dstIdx,
                        const bf16_t* __restrict__ WT,    // [128, 192] (N,K)
                        const float* __restrict__ bias,   // [128]
                        float* __restrict__ aggr,         // [N, 128] zeroed
                        int nEdges)
{
    constexpr int KTOT   = KMSG;
    constexpr int STRIDE = 200;   // 400B rows -> conflict-free b128 reads
    __shared__ __align__(32) bf16_t Alds[TILE_M * STRIDE];
    __shared__ int dlds[TILE_M];

    const int tid = threadIdx.x;
    const int e0  = blockIdx.x * TILE_M;

    // ---- stage A tile: 4 threads per row, 48 elements each -----------------
    {
        const int r  = tid >> 2;
        const int k0 = (tid & 3) * 48;
        const int e  = e0 + r;
        bf16_t* arow = Alds + r * STRIDE;
        if (e < nEdges) {
            const int s = srcIdx[e];
            const float* xrow = x + (long long)s * DFEAT;
            const float* erow = eattr + (long long)e * DEDGE;
            #pragma unroll
            for (int i = 0; i < 48; ++i) {
                int k = k0 + i;
                float v = (k < DFEAT) ? xrow[k] : erow[k - DFEAT];
                arow[k] = (bf16_t)v;
            }
            if ((tid & 3) == 0) dlds[r] = dstIdx[e];
        } else {
            #pragma unroll
            for (int i = 0; i < 48; ++i) arow[k0 + i] = (bf16_t)0.0f;
            if ((tid & 3) == 0) dlds[r] = 0;
        }
    }
    __syncthreads();

    // ---- wave GEMM: each wave = one 16-wide N tile, 4 M tiles --------------
    const int lane = tid & 31;
    const int kh   = lane >> 4;           // half select
    const int nCol = (tid >> 5) * 16 + (lane & 15);
    const bf16_t* wcol = WT + nCol * KTOT;

    v8f acc[4] = {};
    #pragma unroll
    for (int kc = 0; kc < KTOT / 32; ++kc) {
        const int kb = kc * 32;
        // B fragment: lane<16 -> col, K=kb..kb+15 ; lane>=16 -> K=kb+16..kb+31
        v16bf bfrag = *(const v16bf*)(wcol + kb + kh * 16);
        // Batch all four independent A-fragment loads before the WMMAs so the
        // DS traffic overlaps with matrix math instead of forcing dscnt==0.
        AFrag af[4];
        #pragma unroll
        for (int mt = 0; mt < 4; ++mt) {
            // A fragment: elems 0..15 = K base..base+15 (base = kb + kh*16)
            const bf16_t* arow = Alds + (mt * 16 + (lane & 15)) * STRIDE
                                      + kb + kh * 16;
            af[mt].h[0] = *(const v8bf*)(arow);
            af[mt].h[1] = *(const v8bf*)(arow + 8);
        }
        #pragma unroll
        for (int mt = 0; mt < 4; ++mt) {
            acc[mt] = __builtin_amdgcn_wmma_f32_16x16x32_bf16(
                false, af[mt].v, false, bfrag, (short)0, acc[mt], false, false);
        }
    }

    // ---- epilogue: relu + atomic scatter-add -------------------------------
    const float bn = bias[nCol];
    #pragma unroll
    for (int mt = 0; mt < 4; ++mt) {
        #pragma unroll
        for (int r = 0; r < 8; ++r) {
            const int m = mt * 16 + kh * 8 + r;
            if (e0 + m < nEdges) {
                float v = acc[mt][r] + bn;
                v = v > 0.0f ? v : 0.0f;
                atomicAdd(aggr + (long long)dlds[m] * DFEAT + nCol, v);
            }
        }
    }
}

// ---------------------------------------------------------------------------
// Update:  out = relu([aggr , x] @ Wa + ba)    (dense rows = nodes)
// ---------------------------------------------------------------------------
__global__ __launch_bounds__(TPB)
void update_kernel(const float* __restrict__ aggr,  // [N, 128]
                   const float* __restrict__ x,     // [N, 128]
                   const bf16_t* __restrict__ WT,   // [128, 256] (N,K)
                   const float* __restrict__ bias,  // [128]
                   float* __restrict__ out,         // [N, 128]
                   int nNodes)
{
    constexpr int KTOT   = KUPD;
    constexpr int STRIDE = 264;   // 528B rows -> conflict-free b128 reads
    __shared__ __align__(32) bf16_t Alds[TILE_M * STRIDE];

    const int tid   = threadIdx.x;
    const int node0 = blockIdx.x * TILE_M;

    // ---- stage A tile: 4 threads per row, 64 elements each -----------------
    {
        const int r    = tid >> 2;
        const int k0   = (tid & 3) * 64;
        const int node = node0 + r;
        bf16_t* arow = Alds + r * STRIDE;
        if (node < nNodes) {
            const float* srcp = (k0 < DFEAT)
                ? (aggr + (long long)node * DFEAT + k0)
                : (x    + (long long)node * DFEAT + (k0 - DFEAT));
            #pragma unroll
            for (int i = 0; i < 64; ++i) arow[k0 + i] = (bf16_t)srcp[i];
        } else {
            #pragma unroll
            for (int i = 0; i < 64; ++i) arow[k0 + i] = (bf16_t)0.0f;
        }
    }
    __syncthreads();

    const int lane = tid & 31;
    const int kh   = lane >> 4;
    const int nCol = (tid >> 5) * 16 + (lane & 15);
    const bf16_t* wcol = WT + nCol * KTOT;

    v8f acc[4] = {};
    #pragma unroll
    for (int kc = 0; kc < KTOT / 32; ++kc) {
        const int kb = kc * 32;
        v16bf bfrag = *(const v16bf*)(wcol + kb + kh * 16);
        AFrag af[4];
        #pragma unroll
        for (int mt = 0; mt < 4; ++mt) {
            const bf16_t* arow = Alds + (mt * 16 + (lane & 15)) * STRIDE
                                      + kb + kh * 16;
            af[mt].h[0] = *(const v8bf*)(arow);
            af[mt].h[1] = *(const v8bf*)(arow + 8);
        }
        #pragma unroll
        for (int mt = 0; mt < 4; ++mt) {
            acc[mt] = __builtin_amdgcn_wmma_f32_16x16x32_bf16(
                false, af[mt].v, false, bfrag, (short)0, acc[mt], false, false);
        }
    }

    const float bn = bias[nCol];
    #pragma unroll
    for (int mt = 0; mt < 4; ++mt) {
        #pragma unroll
        for (int r = 0; r < 8; ++r) {
            const int m    = mt * 16 + kh * 8 + r;
            const int node = node0 + m;
            if (node < nNodes) {
                float v = acc[mt][r] + bn;
                v = v > 0.0f ? v : 0.0f;
                out[(long long)node * DFEAT + nCol] = v;
            }
        }
    }
}

// ---------------------------------------------------------------------------
// Host driver
// ---------------------------------------------------------------------------
extern "C" void kernel_launch(void* const* d_in, const int* in_sizes, int n_in,
                              void* d_out, int out_size, void* d_ws, size_t ws_size,
                              hipStream_t stream) {
    (void)n_in; (void)out_size; (void)ws_size;

    const float* x     = (const float*)d_in[0];
    const int*   eidx  = (const int*)  d_in[1];   // [2, E] (src row, dst row)
    const float* eattr = (const float*)d_in[2];
    const float* Wm1   = (const float*)d_in[3];
    const float* bm1   = (const float*)d_in[4];
    const float* Wa1   = (const float*)d_in[5];
    const float* ba1   = (const float*)d_in[6];
    const float* Wm2   = (const float*)d_in[7];
    const float* bm2   = (const float*)d_in[8];
    const float* Wa2   = (const float*)d_in[9];
    const float* ba2   = (const float*)d_in[10];

    const int N = in_sizes[0] / DFEAT;     // 50000 nodes
    const int E = in_sizes[1] / 2;         // 640000 edges
    const int* srcIdx = eidx;
    const int* dstIdx = eidx + E;

    // ---- workspace layout --------------------------------------------------
    char* ws = (char*)d_ws;
    size_t off = 0;
    bf16_t* WTm1 = (bf16_t*)(ws + off); off += (size_t)KMSG * DFEAT * sizeof(bf16_t);
    bf16_t* WTa1 = (bf16_t*)(ws + off); off += (size_t)KUPD * DFEAT * sizeof(bf16_t);
    bf16_t* WTm2 = (bf16_t*)(ws + off); off += (size_t)KMSG * DFEAT * sizeof(bf16_t);
    bf16_t* WTa2 = (bf16_t*)(ws + off); off += (size_t)KUPD * DFEAT * sizeof(bf16_t);
    off = (off + 255) & ~(size_t)255;
    float* aggr1 = (float*)(ws + off); off += (size_t)N * DFEAT * sizeof(float);
    float* aggr2 = (float*)(ws + off); off += (size_t)N * DFEAT * sizeof(float);
    float* hbuf  = (float*)(ws + off); off += (size_t)N * DFEAT * sizeof(float);

    // ---- weight prep (bf16, transposed) ------------------------------------
    {
        int tm = (KMSG * DFEAT + TPB - 1) / TPB;
        int ta = (KUPD * DFEAT + TPB - 1) / TPB;
        transpose_w_kernel<<<tm, TPB, 0, stream>>>(Wm1, WTm1, KMSG);
        transpose_w_kernel<<<ta, TPB, 0, stream>>>(Wa1, WTa1, KUPD);
        transpose_w_kernel<<<tm, TPB, 0, stream>>>(Wm2, WTm2, KMSG);
        transpose_w_kernel<<<ta, TPB, 0, stream>>>(Wa2, WTa2, KUPD);
    }

    // ---- zero both aggregation buffers (contiguous) ------------------------
    {
        long long nz = 2LL * N * DFEAT;
        int g = (int)((nz + TPB - 1) / TPB);
        zero_kernel<<<g, TPB, 0, stream>>>(aggr1, nz);
    }

    const int gridE = (E + TILE_M - 1) / TILE_M;
    const int gridN = (N + TILE_M - 1) / TILE_M;

    // ---- layer 1 -----------------------------------------------------------
    msg_scatter_kernel<<<gridE, TPB, 0, stream>>>(x, eattr, srcIdx, dstIdx,
                                                  WTm1, bm1, aggr1, E);
    update_kernel<<<gridN, TPB, 0, stream>>>(aggr1, x, WTa1, ba1, hbuf, N);

    // ---- layer 2 -----------------------------------------------------------
    msg_scatter_kernel<<<gridE, TPB, 0, stream>>>(hbuf, eattr, srcIdx, dstIdx,
                                                  WTm2, bm2, aggr2, E);
    update_kernel<<<gridN, TPB, 0, stream>>>(aggr2, hbuf, WTa2, ba2,
                                             (float*)d_out, N);
}
